// CriticQ_37632503447903
// MI455X (gfx1250) — compile-verified, matrix operand order checked by reading, not run on hardware
//
#include <hip/hip_runtime.h>
#include <hip/hip_bf16.h>

// ---------------------------------------------------------------------------
// Relational critic, fused for MI455X (gfx1250, wave32, WMMA).
//
// Key algebra: Wk(s_j - s_i) = Wk s_j - Wk s_i  => never materialize the
// (B,T,T,K) rel/kk/vv tensors (saves ~0.8 GB of HBM traffic @ 23.3 TB/s).
// MLP stack (the only dense GEMMs, ~0.7 GFLOP) runs on
// v_wmma_f32_16x16x32_bf16 with f32 accumulation.
// ---------------------------------------------------------------------------

#define BT   4096      // B*T
#define KDIM 32
#define TT   256       // T
#define HD   256       // HD1 == HD2

typedef __attribute__((ext_vector_type(16))) __bf16 v16bf;
typedef __attribute__((ext_vector_type(8)))  __bf16 v8bf;
typedef __attribute__((ext_vector_type(8)))  float  v8f;

// Load a 16x32 bf16 WMMA fragment from a row-major matrix (ld = row length).
// ISA 16-bit A/B layout: lane L -> row (L&15); per-lane 16 elements cover
// K = k0 + half*8 + e (e<8) and K = k0 + 16 + half*8 + (e-8), half = L>>4.
// Each half is 8 contiguous bf16 -> one 16-byte load.
__device__ __forceinline__ v16bf load_frag_bf16(const __bf16* __restrict__ base,
                                                int ld, int row0, int k0, int lane) {
    int half = lane >> 4;
    int r    = lane & 15;
    const __bf16* p = base + (size_t)(row0 + r) * ld + k0 + half * 8;
    v8bf lo = *(const v8bf*)(p);
    v8bf hi = *(const v8bf*)(p + 16);
    return __builtin_shufflevector(lo, hi, 0,1,2,3,4,5,6,7,8,9,10,11,12,13,14,15);
}

// ---------------------------------------------------------------------------
// K0: convert W1 (256x64) and W2 (256x256) to bf16
// ---------------------------------------------------------------------------
__global__ void cvt_weights_kernel(const float* __restrict__ W1,
                                   const float* __restrict__ W2,
                                   __bf16* __restrict__ w1b,
                                   __bf16* __restrict__ w2b) {
    int i = blockIdx.x * blockDim.x + threadIdx.x;   // 65536 threads
    if (i < HD * 64)  w1b[i] = (__bf16)W1[i];
    if (i < HD * HD)  w2b[i] = (__bf16)W2[i];
}

// ---------------------------------------------------------------------------
// K1: per-row projections.  row = b*T + t, one wave per row.
//   s = concat(state, action)                 (kept in LDS, bf16 copy -> Xin)
//   q  = relu(Wq s + bq),  ks = Wk s,  vs = Wv s   (bias/relu for k,v applied
//                                                    per-pair in attention)
// ---------------------------------------------------------------------------
__global__ void proj_kernel(const float* __restrict__ state,
                            const float* __restrict__ action,
                            const float* __restrict__ Wq, const float* __restrict__ bq,
                            const float* __restrict__ Wk, const float* __restrict__ Wv,
                            float* __restrict__ q, float* __restrict__ ks,
                            float* __restrict__ vs, __bf16* __restrict__ Xin) {
    __shared__ float sh_s[8][KDIM];
    int wave = threadIdx.x >> 5;
    int lane = threadIdx.x & 31;
    int row  = blockIdx.x * 8 + wave;

    float sv = (lane < 24) ? state[(size_t)row * 24 + lane]
                           : action[(size_t)row * 8 + (lane - 24)];
    sh_s[wave][lane] = sv;
    __syncthreads();

    const float* wqr = Wq + lane * KDIM;
    const float* wkr = Wk + lane * KDIM;
    const float* wvr = Wv + lane * KDIM;
    float aq = bq[lane], ak = 0.f, av = 0.f;
#pragma unroll
    for (int c = 0; c < KDIM; ++c) {
        float x = sh_s[wave][c];
        aq = fmaf(x, wqr[c], aq);
        ak = fmaf(x, wkr[c], ak);
        av = fmaf(x, wvr[c], av);
    }
    q [(size_t)row * KDIM + lane] = fmaxf(aq, 0.f);
    ks[(size_t)row * KDIM + lane] = ak;
    vs[(size_t)row * KDIM + lane] = av;
    // second half of GEMM-1 input: bf16 copy of s
    Xin[(size_t)row * 64 + KDIM + lane] = (__bf16)sh_s[wave][lane];
}

// ---------------------------------------------------------------------------
// K2: attention for one (b,i) per block (256 threads).
//   scores[j] = sum_k relu(ks_j - ks_i + bk) * q_i / sqrt(32)
//   w = softmax_j(scores)
//   x[k] = sum_j w_j * relu(vs_j - vs_i + bv)        -> Xin[row, 0:32] (bf16)
// ---------------------------------------------------------------------------
__global__ void attn_kernel(const float* __restrict__ q,
                            const float* __restrict__ ks,
                            const float* __restrict__ vs,
                            const float* __restrict__ bk,
                            const float* __restrict__ bv,
                            __bf16* __restrict__ Xin) {
    __shared__ float sh_ksi[KDIM], sh_q[KDIM], sh_vsi[KDIM], sh_bk[KDIM], sh_bv[KDIM];
    __shared__ float sh_w[TT];
    __shared__ float red[TT];

    int t   = threadIdx.x;           // 0..255  (= j in phase 1)
    int row = blockIdx.x;            // b*T + i
    int b   = row >> 8;

    if (t < KDIM) {
        sh_ksi[t] = ks[(size_t)row * KDIM + t];
        sh_q[t]   = q [(size_t)row * KDIM + t];
        sh_vsi[t] = vs[(size_t)row * KDIM + t];
        sh_bk[t]  = bk[t];
        sh_bv[t]  = bv[t];
    }
    __syncthreads();

    // ---- scores -----------------------------------------------------------
    const float* ksj = ks + (size_t)(b * TT + t) * KDIM;
    float acc = 0.f;
#pragma unroll
    for (int k = 0; k < KDIM; ++k)
        acc += fmaxf(ksj[k] - sh_ksi[k] + sh_bk[k], 0.f) * sh_q[k];
    float sc = acc * 0.17677669529663687f;   // 1/sqrt(32)

    // ---- softmax over the 256 j's ----------------------------------------
    red[t] = sc; __syncthreads();
    for (int off = 128; off > 0; off >>= 1) {
        if (t < off) red[t] = fmaxf(red[t], red[t + off]);
        __syncthreads();
    }
    float mx = red[0]; __syncthreads();
    float e = __expf(sc - mx);
    red[t] = e; __syncthreads();
    for (int off = 128; off > 0; off >>= 1) {
        if (t < off) red[t] += red[t + off];
        __syncthreads();
    }
    float inv = 1.f / red[0];
    sh_w[t] = e * inv;
    __syncthreads();

    // ---- weighted relu-vv sum:  lane layout (g = j-group, k = channel) ----
    int k = t & 31, g = t >> 5;
    float vik = sh_vsi[k], bvk = sh_bv[k];
    float xa = 0.f;
    for (int j = g; j < TT; j += 8)
        xa += sh_w[j] * fmaxf(vs[(size_t)(b * TT + j) * KDIM + k] - vik + bvk, 0.f);
    red[t] = xa;
    __syncthreads();
    if (g == 0) {
        float s = 0.f;
#pragma unroll
        for (int gg = 0; gg < 8; ++gg) s += red[gg * 32 + k];
        Xin[(size_t)row * 64 + k] = (__bf16)s;
    }
}

// ---------------------------------------------------------------------------
// K3/K4: WMMA GEMM  Y = act(X @ W^T + bias),  X: (4096 x KD) bf16 row-major,
// W: (256 x KD) bf16 row-major.  One wave per 16x16 output tile.
// grid = (M/16, 2), block = 256 (8 waves); wave w -> col tile blockIdx.y*8+w.
// ---------------------------------------------------------------------------
template <int KD, bool RELU, typename OUT_T>
__global__ void gemm_wmma_kernel(const __bf16* __restrict__ X,
                                 const __bf16* __restrict__ W,
                                 const float* __restrict__ bias,
                                 OUT_T* __restrict__ Y) {
    int lane = threadIdx.x & 31;
    int wave = threadIdx.x >> 5;
    int row0 = blockIdx.x * 16;
    int col0 = (blockIdx.y * 8 + wave) * 16;

    v8f acc = {0.f, 0.f, 0.f, 0.f, 0.f, 0.f, 0.f, 0.f};
#pragma unroll
    for (int k0 = 0; k0 < KD; k0 += 32) {
        v16bf a = load_frag_bf16(X, KD, row0, k0, lane);
        v16bf b = load_frag_bf16(W, KD, col0, k0, lane);
        // (neg_a, A, neg_b, B, c_mod, C, reuse_a, reuse_b)
        acc = __builtin_amdgcn_wmma_f32_16x16x32_bf16(false, a, false, b,
                                                      (short)0, acc, false, false);
    }

    // D layout: VGPR v, lane L -> M = v + 8*(L>>4), N = L&15
    int half = lane >> 4, n = lane & 15;
    float bn = bias[col0 + n];
#pragma unroll
    for (int v = 0; v < 8; ++v) {
        int m = v + 8 * half;
        float val = acc[v] + bn;
        if (RELU) val = fmaxf(val, 0.f);
        Y[(size_t)(row0 + m) * HD + (col0 + n)] = (OUT_T)val;
    }
}

// ---------------------------------------------------------------------------
// K5: final GEMV  value[row] = Y2[row,:] . Wout + bout.  One wave per row.
// ---------------------------------------------------------------------------
__global__ void out_gemv_kernel(const float* __restrict__ Y2,
                                const float* __restrict__ Wout,
                                const float* __restrict__ bout,
                                float* __restrict__ out) {
    int wave = threadIdx.x >> 5;
    int lane = threadIdx.x & 31;
    int row  = blockIdx.x * 8 + wave;
    const float* yr = Y2 + (size_t)row * HD;
    float acc = 0.f;
#pragma unroll
    for (int c = lane; c < HD; c += 32) acc += yr[c] * Wout[c];
#pragma unroll
    for (int off = 16; off > 0; off >>= 1) acc += __shfl_xor(acc, off, 32);
    if (lane == 0) out[row] = acc + bout[0];
}

// ---------------------------------------------------------------------------
extern "C" void kernel_launch(void* const* d_in, const int* in_sizes, int n_in,
                              void* d_out, int out_size, void* d_ws, size_t ws_size,
                              hipStream_t stream) {
    (void)in_sizes; (void)n_in; (void)out_size; (void)ws_size;

    const float* state  = (const float*)d_in[0];
    const float* action = (const float*)d_in[1];
    const float* Wq     = (const float*)d_in[2];
    const float* bq     = (const float*)d_in[3];
    const float* Wk     = (const float*)d_in[4];
    const float* bk     = (const float*)d_in[5];
    const float* Wv     = (const float*)d_in[6];
    const float* bv     = (const float*)d_in[7];
    const float* W1     = (const float*)d_in[8];
    const float* b1     = (const float*)d_in[9];
    const float* W2     = (const float*)d_in[10];
    const float* b2     = (const float*)d_in[11];
    const float* Wout   = (const float*)d_in[12];
    const float* bout   = (const float*)d_in[13];
    float* out = (float*)d_out;

    // workspace carving (all slice sizes are multiples of 256 bytes)
    char* ws = (char*)d_ws;
    float*  q_buf  = (float*)ws;            ws += (size_t)BT * KDIM * 4;   // 512 KB
    float*  ks_buf = (float*)ws;            ws += (size_t)BT * KDIM * 4;   // 512 KB
    float*  vs_buf = (float*)ws;            ws += (size_t)BT * KDIM * 4;   // 512 KB
    __bf16* Xin    = (__bf16*)ws;           ws += (size_t)BT * 64 * 2;     // 512 KB
    __bf16* w1b    = (__bf16*)ws;           ws += (size_t)HD * 64 * 2;     //  32 KB
    __bf16* w2b    = (__bf16*)ws;           ws += (size_t)HD * HD * 2;     // 128 KB
    __bf16* Y1     = (__bf16*)ws;           ws += (size_t)BT * HD * 2;     //   2 MB
    float*  Y2     = (float*)ws;            ws += (size_t)BT * HD * 4;     //   4 MB

    // K0: weight conversion (covers max(16384, 65536) elements)
    cvt_weights_kernel<<<256, 256, 0, stream>>>(W1, W2, w1b, w2b);

    // K1: projections (8 rows/block -> 512 blocks)
    proj_kernel<<<BT / 8, 256, 0, stream>>>(state, action, Wq, bq, Wk, Wv,
                                            q_buf, ks_buf, vs_buf, Xin);

    // K2: attention, one block per (b,i)
    attn_kernel<<<BT, 256, 0, stream>>>(q_buf, ks_buf, vs_buf, bk, bv, Xin);

    // K3: Y1 = relu(Xin @ W1^T + b1)   (K = 64)
    gemm_wmma_kernel<64, true, __bf16><<<dim3(BT / 16, 2), 256, 0, stream>>>(
        Xin, w1b, b1, Y1);

    // K4: Y2 = relu(Y1 @ W2^T + b2)    (K = 256)
    gemm_wmma_kernel<256, true, float><<<dim3(BT / 16, 2), 256, 0, stream>>>(
        Y1, w2b, b2, Y2);

    // K5: value = Y2 @ Wout^T + bout
    out_gemv_kernel<<<BT / 8, 256, 0, stream>>>(Y2, Wout, bout, out);
}
